// Attention_19791209300650
// MI455X (gfx1250) — matrix operand order repaired and checked by reading, not compiled, not run on hardware
//
#include <hip/hip_runtime.h>
#include <math.h>

typedef __attribute__((ext_vector_type(16))) _Float16 v16h;
typedef __attribute__((ext_vector_type(8)))  float    v8f;
typedef __attribute__((ext_vector_type(4)))  unsigned int v4u;
typedef __attribute__((ext_vector_type(8)))  int      v8i_;
typedef __attribute__((ext_vector_type(4)))  int      v4i_;

#ifndef __has_builtin
#define __has_builtin(x) 0
#endif
#if __has_builtin(__builtin_amdgcn_tensor_load_to_lds) && \
    __has_builtin(__builtin_amdgcn_s_wait_tensorcnt)
#define HAVE_TDM 1
#else
#define HAVE_TDM 0
#endif

#define LQn 512
#define LKn 512
#define Dn  64
#define HIDn 128
#define Bn  2

// ws layout (float offsets)
#define WS_QP   0
#define WS_KP   65536
#define WS_VP   131072
#define WS_QA   196608
#define WS_QB   327680
#define WS_KA   458752
#define WS_KB   589824
#define WS_OUTP 720896

__device__ __forceinline__ v8f wmma16(v16h a, v16h b, v8f c) {
  return __builtin_amdgcn_wmma_f32_16x16x32_f16(false, a, false, b, (short)0, c,
                                                false, false);
}

// A fragment: 16x32 f16 tile from row-major f32 X (leading dim ld), K offset k0.
__device__ __forceinline__ v16h load_a(const float* X, int ld, int lane, int k0) {
  const int m  = lane & 15;
  const int hf = lane >> 4;
  const float* row = X + m * ld + k0 + hf * 8;
  v16h a;
#pragma unroll
  for (int vv = 0; vv < 8; ++vv) {
    const int kb = ((vv >> 2) << 4) + ((vv & 3) << 1);
    a[2 * vv]     = (_Float16)row[kb];
    a[2 * vv + 1] = (_Float16)row[kb + 1];
  }
  return a;
}

// B fragment: 32x16 f16 tile from row-major f32 W (leading dim ld): lane%16 = N, elems = K (table order).
__device__ __forceinline__ v16h load_b(const float* W, int ld, int lane, int k0, int n0) {
  const int n  = lane & 15;
  const int hf = lane >> 4;
  const float* col = W + (k0 + hf * 8) * ld + n0 + n;
  v16h b;
#pragma unroll
  for (int e = 0; e < 16; ++e) b[e] = (_Float16)col[((e >> 3) * 16 + (e & 7)) * ld];
  return b;
}

#if HAVE_TDM
// TDM: DMA a 16 x 128-float (16 x 512B) row-major tile from global into LDS.
// D# per CDNA5 ISA: group0 = {count|flags, lds_addr,
// global_addr[31:0], global_addr[56:32] | type=2<<30}; group1 packs data_size=4B,
// tensor_dim0=128, tensor_dim1=512, tile_dim0=128, tile_dim1=16, dim0_stride=128.
// clang-23 toolchain: 6-arg builtin (g0, g1, g2, g3, g4, cpol).
__device__ __forceinline__ void tdm_load_tile16x128(const float* gsrc,
                                                    unsigned int lds_byte) {
  const unsigned long long ga = (unsigned long long)(uintptr_t)gsrc;
  v4u g0;
  g0[0] = 1u;                                     // count=1, is_restore=0
  g0[1] = lds_byte;                               // lds_addr
  g0[2] = (unsigned int)ga;                       // global_addr[31:0]
  g0[3] = ((unsigned int)(ga >> 32) & 0x01FFFFFFu) | 0x80000000u; // [56:32]|type=2
  v8i_ g1;
  g1[0] = 0x00020000;          // workgroup_mask=0, data_size=2 (4 bytes)
  g1[1] = HIDn << 16;          // tensor_dim0[15:0]=128 in [31:16]
  g1[2] = 512 << 16;           // tensor_dim0[31:16]=0, tensor_dim1[15:0]=512
  g1[3] = HIDn << 16;          // tensor_dim1[31:16]=0, tile_dim0=128
  g1[4] = 16;                  // tile_dim1=16, tile_dim2=0
  g1[5] = HIDn;                // tensor_dim0_stride[31:0]=128
  g1[6] = 0;                   // stride hi / dim1_stride lo
  g1[7] = 0;                   // dim1_stride hi
  const v4i_ z4 = {0, 0, 0, 0};
  const v8i_ z8 = {0, 0, 0, 0, 0, 0, 0, 0};
  __builtin_amdgcn_tensor_load_to_lds(g0, g1, z4, z4, z8, 0);
}
#endif

// ---------------- Kernel 1: projections + additive-attention features ----------
__global__ void k_proj(const float* __restrict__ q, const float* __restrict__ k,
                       const float* __restrict__ v, const float* __restrict__ Ww,
                       const float* __restrict__ wb, const float* __restrict__ W1,
                       const float* __restrict__ b1, float* __restrict__ ws) {
  __shared__ float plds[4][16 * Dn];
  const int lane = threadIdx.x & 31;
  const int widx = threadIdx.x >> 5;
  const int gw   = blockIdx.x * 4 + widx;       // 0..191
  const int mat  = gw / 64;
  const int r    = gw % 64;
  const int b    = r >> 5;
  const int row0 = (r & 31) * 16;
  const float* X = (mat == 0 ? q : (mat == 1 ? k : v)) + (b * LQn + row0) * Dn;

  // stage 1: P = X @ Ww + wb  (16x64)
  v8f cacc[4] = {};
#pragma unroll
  for (int ks = 0; ks < Dn; ks += 32) {
    v16h a = load_a(X, Dn, lane, ks);
#pragma unroll
    for (int nt = 0; nt < 4; ++nt) {
      v16h bf = load_b(Ww, Dn, lane, ks, nt * 16);
      cacc[nt] = wmma16(a, bf, cacc[nt]);
    }
  }
  const int n  = lane & 15;
  const int hf = lane >> 4;
  float* P = ws + (mat == 0 ? WS_QP : (mat == 1 ? WS_KP : WS_VP)) +
             (b * LQn + row0) * Dn;
#pragma unroll
  for (int nt = 0; nt < 4; ++nt) {
    const float bias = wb[nt * 16 + n];
#pragma unroll
    for (int vv = 0; vv < 8; ++vv) {
      const float val = cacc[nt][vv] + bias;
      P[(vv + 8 * hf) * Dn + nt * 16 + n]          = val;
      plds[widx][(vv + 8 * hf) * Dn + nt * 16 + n] = val;
    }
  }
  if (mat == 2) return;

  // same-wave LDS RAW (cross-lane): make DS writes visible before fragment reads
  asm volatile("s_wait_dscnt 0" ::: "memory");

  // stage 2: P @ W1q -> (qa|ka), P @ W1k -> (qb|kb); fold b1 into qa,qb
  const float* T = plds[widx];
  float* dst1 = ws + (mat == 0 ? WS_QA : WS_KA) + (b * LQn + row0) * HIDn;
  float* dst2 = ws + (mat == 0 ? WS_QB : WS_KB) + (b * LQn + row0) * HIDn;
  for (int sel = 0; sel < 2; ++sel) {
    const float* W = W1 + (sel ? Dn * HIDn : 0);
    float* dst = sel ? dst2 : dst1;
    v8f acc[8] = {};
#pragma unroll
    for (int ks = 0; ks < Dn; ks += 32) {
      v16h a = load_a(T, Dn, lane, ks);
#pragma unroll
      for (int nt = 0; nt < 8; ++nt) {
        v16h bf = load_b(W, HIDn, lane, ks, nt * 16);
        acc[nt] = wmma16(a, bf, acc[nt]);
      }
    }
#pragma unroll
    for (int nt = 0; nt < 8; ++nt) {
      const float bias = (mat == 0) ? b1[nt * 16 + n] : 0.0f;
#pragma unroll
      for (int vv = 0; vv < 8; ++vv)
        dst[(vv + 8 * hf) * HIDn + nt * 16 + n] = acc[nt][vv] + bias;
    }
  }
}

// ---------------- Kernel 2: pairwise relu-MLP scores + softmax + attn@vp -------
__global__ void k_attn(const float* __restrict__ mask, const float* __restrict__ W2,
                       const float* __restrict__ b2, float* __restrict__ ws,
                       float* __restrict__ d_out) {
  __shared__ float sQa[16][HIDn];
  __shared__ float sQb[16][HIDn];
  __shared__ float sKa[16][HIDn];
  __shared__ float sKb[16][HIDn];
  __shared__ float sL[16][LKn];          // total static LDS = 64 KB

  const int tid = threadIdx.x;
  const int b   = blockIdx.x >> 5;
  const int i0  = (blockIdx.x & 31) * 16;
  const int lane = tid & 31;
  const int w    = tid >> 5;

  const float* qa = ws + WS_QA + (b * LQn + i0) * HIDn;
  const float* qb = ws + WS_QB + (b * LQn + i0) * HIDn;
  const float* ka = ws + WS_KA + b * LKn * HIDn;
  const float* kb = ws + WS_KB + b * LKn * HIDn;
  const float* vp = ws + WS_VP + b * LKn * Dn;
  float* outp     = ws + WS_OUTP + (b * LQn + i0) * Dn;
  float* attn_out = d_out + Bn * LQn * Dn + (b * LQn + i0) * LKn;

  for (int t = tid; t < 16 * HIDn; t += 256) {
    sQa[t >> 7][t & 127] = qa[t];
    sQb[t >> 7][t & 127] = qb[t];
  }
  // prefetch vp rows for the final attn@vp GEMM (global_prefetch_b8)
  __builtin_prefetch(vp + tid * 2 * Dn, 0, 1);
  __syncthreads();

  const float two_b2 = 2.0f * b2[0];
  const int li = tid & 15;    // query row in tile
  const int lj = tid >> 4;    // key row in chunk

  for (int j0 = 0; j0 < LKn; j0 += 16) {
    __syncthreads();
#if HAVE_TDM
    if (w == 0) {   // wave 0 issues two 16x512B tile DMAs via the Tensor Data Mover
      tdm_load_tile16x128(ka + j0 * HIDn, (unsigned int)(uintptr_t)&sKa[0][0]);
      tdm_load_tile16x128(kb + j0 * HIDn, (unsigned int)(uintptr_t)&sKb[0][0]);
      __builtin_amdgcn_s_wait_tensorcnt(0);
    }
#else
    for (int t = tid; t < 16 * HIDn; t += 256) {
      sKa[t >> 7][t & 127] = ka[j0 * HIDn + t];
      sKb[t >> 7][t & 127] = kb[j0 * HIDn + t];
    }
    if (j0 + 16 < LKn) {
      __builtin_prefetch(ka + (j0 + 16) * HIDn + tid, 0, 1);
      __builtin_prefetch(kb + (j0 + 16) * HIDn + tid, 0, 1);
    }
#endif
    __syncthreads();

    float s = two_b2;
#pragma unroll
    for (int h = 0; h < HIDn; h += 4) {
      const float4 a1 = *(const float4*)&sQa[li][h];
      const float4 k1 = *(const float4*)&sKb[lj][h];
      const float4 a2 = *(const float4*)&sQb[li][h];
      const float4 k2 = *(const float4*)&sKa[lj][h];
      const float4 wv = *(const float4*)&W2[h];        // uniform -> scalar loads
      s += (fmaxf(a1.x + k1.x, 0.f) + fmaxf(a2.x + k2.x, 0.f)) * wv.x;
      s += (fmaxf(a1.y + k1.y, 0.f) + fmaxf(a2.y + k2.y, 0.f)) * wv.y;
      s += (fmaxf(a1.z + k1.z, 0.f) + fmaxf(a2.z + k2.z, 0.f)) * wv.z;
      s += (fmaxf(a1.w + k1.w, 0.f) + fmaxf(a2.w + k2.w, 0.f)) * wv.w;
    }
    s += mask[(b * LQn + i0 + li) * LKn + j0 + lj] * -1e9f;
    sL[li][j0 + lj] = s;
  }
  __syncthreads();

  // softmax: 8 waves x 2 rows; each 16-lane half-wave owns one row of 512
  {
    const int rr = 2 * w + (lane >> 4);
    const int lr = lane & 15;
    float m = -3.0e38f;
    for (int c = lr; c < LKn; c += 16) m = fmaxf(m, sL[rr][c]);
#pragma unroll
    for (int off = 8; off; off >>= 1) m = fmaxf(m, __shfl_xor(m, off, 16));
    float sum = 0.0f;
    for (int c = lr; c < LKn; c += 16) {
      float e = __expf(sL[rr][c] - m);
      sL[rr][c] = e;
      sum += e;
    }
#pragma unroll
    for (int off = 8; off; off >>= 1) sum += __shfl_xor(sum, off, 16);
    const float inv = 1.0f / sum;
    for (int c = lr; c < LKn; c += 16) {
      const float a = sL[rr][c] * inv;
      sL[rr][c] = a;
      attn_out[rr * LKn + c] = a;
    }
  }
  __syncthreads();

  // attn @ vp : waves 0..3 each produce one 16x16 column tile of the 16x64 output
  if (w < 4) {
    v8f acc = {};
    for (int ks = 0; ks < LKn; ks += 32) {
      v16h a  = load_a(&sL[0][0], LKn, lane, ks);
      v16h bf = load_b(vp, Dn, lane, ks, w * 16);
      acc = wmma16(a, bf, acc);
    }
    const int nn = lane & 15, hh = lane >> 4;
#pragma unroll
    for (int vv = 0; vv < 8; ++vv)
      outp[(vv + 8 * hh) * Dn + w * 16 + nn] = acc[vv];
  }
}

// ---------------- Kernel 3: out = outp @ Wd + db ------------------------------
__global__ void k_outproj(const float* __restrict__ Wd, const float* __restrict__ db,
                          const float* __restrict__ ws, float* __restrict__ d_out) {
  const int lane = threadIdx.x & 31;
  const int gw   = blockIdx.x * 4 + (threadIdx.x >> 5);   // 0..63
  const int b    = gw >> 5;
  const int row0 = (gw & 31) * 16;
  const float* X = ws + WS_OUTP + (b * LQn + row0) * Dn;
  float* Y       = d_out + (b * LQn + row0) * Dn;

  v8f acc[4] = {};
#pragma unroll
  for (int ks = 0; ks < Dn; ks += 32) {
    v16h a = load_a(X, Dn, lane, ks);
#pragma unroll
    for (int nt = 0; nt < 4; ++nt) {
      v16h bf = load_b(Wd, Dn, lane, ks, nt * 16);
      acc[nt] = wmma16(a, bf, acc[nt]);
    }
  }
  const int n = lane & 15, hf = lane >> 4;
#pragma unroll
  for (int nt = 0; nt < 4; ++nt) {
    const float bias = db[nt * 16 + n];
#pragma unroll
    for (int vv = 0; vv < 8; ++vv)
      Y[(vv + 8 * hf) * Dn + nt * 16 + n] = acc[nt][vv] + bias;
  }
}

extern "C" void kernel_launch(void* const* d_in, const int* in_sizes, int n_in,
                              void* d_out, int out_size, void* d_ws, size_t ws_size,
                              hipStream_t stream) {
  const float* q    = (const float*)d_in[0];
  const float* k    = (const float*)d_in[1];
  const float* v    = (const float*)d_in[2];
  const float* mask = (const float*)d_in[3];
  const float* Ww   = (const float*)d_in[4];
  const float* wb   = (const float*)d_in[5];
  const float* Wd   = (const float*)d_in[6];
  const float* db   = (const float*)d_in[7];
  const float* W1   = (const float*)d_in[8];
  const float* b1   = (const float*)d_in[9];
  const float* W2   = (const float*)d_in[10];
  const float* b2   = (const float*)d_in[11];
  float* ws  = (float*)d_ws;
  float* out = (float*)d_out;

  k_proj<<<48, 128, 0, stream>>>(q, k, v, Ww, wb, W1, b1, ws);
  k_attn<<<64, 256, 0, stream>>>(mask, W2, b2, ws, out);
  k_outproj<<<16, 128, 0, stream>>>(Wd, db, ws, out);
}
